// Attention_2654289789382
// MI455X (gfx1250) — matrix operand order repaired and checked by reading, not compiled, not run on hardware
//
#include <hip/hip_runtime.h>

#define B_     4
#define N_     2048
#define D_     512
#define H_     8
#define DH_    64
#define INNER_ 512
#define M_     (B_ * N_)   // 8192

typedef __attribute__((ext_vector_type(16))) __bf16 v16bf;
typedef __attribute__((ext_vector_type(8)))  __bf16 v8bf;
typedef __attribute__((ext_vector_type(8)))  float  v8f;

#if __has_builtin(__builtin_amdgcn_tensor_load_to_lds) && __has_builtin(__builtin_amdgcn_s_wait_tensorcnt)
#define USE_TDM 1
#else
#define USE_TDM 0
#endif

#define STR2(x) #x
#define STR(x) STR2(x)
#pragma message("probe: clang_major=" STR(__clang_major__))
#if USE_TDM
#pragma message("probe: USE_TDM=1 (tensor_load_to_lds builtin present)")
#else
#pragma message("probe: USE_TDM=0 (tensor_load_to_lds builtin absent)")
#endif

#if USE_TDM
typedef __attribute__((ext_vector_type(4))) unsigned int u32x4;
typedef __attribute__((ext_vector_type(4))) int          i32x4;
typedef __attribute__((ext_vector_type(8))) int          i32x8;

// LDS byte offset of a __shared__ object (allocation-relative address).
#define LDS_OFF(p) ((unsigned)(unsigned long long)(__attribute__((address_space(3))) const void*)(p))

// Issue a TDM load of a (rows x width) bf16 tile with row stride strideElems
// into LDS at ldsAddr (tile stored dense row-major: width*2 bytes per row).
static __device__ inline void tdm_issue_2d_bf16(const __bf16* gsrc, unsigned ldsAddr,
                                                unsigned width, unsigned rows,
                                                unsigned strideElems) {
  unsigned long long ga = (unsigned long long)(__UINTPTR_TYPE__)gsrc;
  u32x4 g0;
  g0[0] = 1u;                                   // count=1, user descriptor
  g0[1] = ldsAddr;                              // lds_addr (bytes)
  g0[2] = (unsigned)ga;                         // global_addr[31:0]
  g0[3] = (unsigned)((ga >> 32) & 0x01FFFFFFu)  // global_addr[56:32]
          | 0x80000000u;                        // type=2 ("image")
  i32x8 g1;
  g1[0] = 0x00010000;                           // data_size=1 (2 bytes)
  g1[1] = (int)(width << 16);                   // tensor_dim0 lo16 (bits 63:48)
  g1[2] = (int)(rows << 16);                    // dim0 hi=0 | tensor_dim1 lo16
  g1[3] = (int)(width << 16);                   // dim1 hi=0 | tile_dim0
  g1[4] = (int)rows;                            // tile_dim1, tile_dim2=0
  g1[5] = (int)strideElems;                     // tensor_dim0_stride lo32
  g1[6] = 0;                                    // stride hi, dim1_stride lo
  g1[7] = 0;
  i32x4 z4 = {};
#if __clang_major__ >= 23
  i32x8 z8 = {};
  __builtin_amdgcn_tensor_load_to_lds(g0, g1, z4, z4, z8, 0);
#else
  __builtin_amdgcn_tensor_load_to_lds(g0, g1, z4, z4, 0);
#endif
}
#endif  // USE_TDM

// ---------------------------------------------------------------------------
// Fragment loaders matching the CDNA5 WMMA VGPR layouts (CDNA5 ISA)
// ---------------------------------------------------------------------------
static __device__ inline v16bf frag_a_16x32(const __bf16* tile, int ld) {
  const int l = threadIdx.x & 31;
  const int row = l & 15, half = l >> 4;
  const __bf16* p = tile + row * ld + half * 8;
  v8bf lo = *(const v8bf*)p;
  v8bf hi = *(const v8bf*)(p + 16);
  v16bf f;
#pragma unroll
  for (int e = 0; e < 8; ++e) { f[e] = lo[e]; f[e + 8] = hi[e]; }
  return f;
}

static __device__ inline v16bf frag_b_32x16(const __bf16* tile, int ld, int n0) {
  const int l = threadIdx.x & 31;           // N = lane&15, K in table order
  const int half = l >> 4;
  const __bf16* p = tile + n0 + (l & 15);
  v16bf f;
#pragma unroll
  for (int e = 0; e < 8; ++e) { f[e] = p[(half * 8 + e) * ld]; f[e + 8] = p[(16 + half * 8 + e) * ld]; }
  return f;
}

// ---------------------------------------------------------------------------
__global__ __launch_bounds__(256) void cvt_bf16_kernel(const float* __restrict__ in,
                                                       __bf16* __restrict__ out, int n) {
  int i = blockIdx.x * 256 + threadIdx.x;
  if (i < n) out[i] = (__bf16)in[i];
}

// ---------------------------------------------------------------------------
// sigma: one wave per (b,i) row. r[b,h,i] = 1 / (exp(sigmoid(x.Ws + bs)) + 1)
// ---------------------------------------------------------------------------
__global__ __launch_bounds__(256) void sigma_kernel(const float* __restrict__ x,
                                                    const float* __restrict__ Ws,
                                                    const float* __restrict__ bs,
                                                    float* __restrict__ rinv) {
  const int wid = threadIdx.x >> 5, lane = threadIdx.x & 31;
  const int m = blockIdx.x * 8 + wid;
  const int bb = m >> 11, i = m & (N_ - 1);
  const float* xr = x + (size_t)m * D_;
  float acc[H_];
#pragma unroll
  for (int h = 0; h < H_; ++h) acc[h] = 0.0f;
  for (int k = lane; k < D_; k += 32) {
    const float xv = xr[k];
    const float* wr = Ws + k * H_;
#pragma unroll
    for (int h = 0; h < H_; ++h) acc[h] += xv * wr[h];
  }
#pragma unroll
  for (int off = 16; off >= 1; off >>= 1)
#pragma unroll
    for (int h = 0; h < H_; ++h) acc[h] += __shfl_xor(acc[h], off, 32);
  if (lane == 0) {
#pragma unroll
    for (int h = 0; h < H_; ++h) {
      float z = acc[h] + bs[h];
      float sig = 1.0f / (1.0f + __expf(-z));
      rinv[((size_t)bb * H_ + h) * N_ + i] = 1.0f / (__expf(sig) + 1.0f);
    }
  }
}

// ---------------------------------------------------------------------------
// bf16 WMMA GEMM: C[M,N] = A[M,K]*B[K,N] (+bias). Block tile 128x64, 8 waves.
// Tiles arrive via TDM (double-buffered, s_wait_tensorcnt) when available,
// else register-staged ping-pong copies. One barrier per K-step either way.
// ---------------------------------------------------------------------------
__global__ __launch_bounds__(256) void gemm_bf16_wmma(const __bf16* __restrict__ A,
                                                      const __bf16* __restrict__ Bm,
                                                      __bf16* __restrict__ outBf,
                                                      float* __restrict__ outF,
                                                      const float* __restrict__ bias,
                                                      int M, int N, int K) {
  __shared__ __align__(16) __bf16 ldsA[2][128 * 32];
  __shared__ __align__(16) __bf16 ldsB[2][32 * 64];
  const int t = threadIdx.x;
  const int wid = t >> 5, lane = t & 31;
  const int mBase = blockIdx.x * 128;
  const int nBase = blockIdx.y * 64;

  v8f acc[4] = {};

#if USE_TDM
  const __bf16* gAt = A + (size_t)mBase * K;        // tile origin (row mBase, col 0)
  const __bf16* gBt = Bm + nBase;                   // tile origin (row 0, col nBase)
  if (wid == 0) {
    tdm_issue_2d_bf16(gAt, LDS_OFF(&ldsA[0][0]), 32, 128, (unsigned)K);
    tdm_issue_2d_bf16(gBt, LDS_OFF(&ldsB[0][0]), 64, 32, (unsigned)N);
  }
#else
  const int arow = t >> 1, acol = (t & 1) * 16;   // 128x32 tile: 16 elems/thread
  const int brow = t >> 3, bcol = (t & 7) * 8;    // 32x64 tile:   8 elems/thread
  const __bf16* gA = A + (size_t)(mBase + arow) * K + acol;
  const __bf16* gB = Bm + (size_t)brow * N + nBase + bcol;
  v8bf ra0 = *(const v8bf*)gA;
  v8bf ra1 = *(const v8bf*)(gA + 8);
  v8bf rb0 = *(const v8bf*)gB;
#endif

  for (int k0 = 0; k0 < K; k0 += 32) {
    const int buf = (k0 >> 5) & 1;
#if USE_TDM
    if (wid == 0) __builtin_amdgcn_s_wait_tensorcnt(0);   // current tiles landed
    __syncthreads();                                      // visible to all waves
    if (wid == 0 && k0 + 32 < K) {                        // prefetch next tiles
      tdm_issue_2d_bf16(gAt + k0 + 32, LDS_OFF(&ldsA[buf ^ 1][0]), 32, 128, (unsigned)K);
      tdm_issue_2d_bf16(gBt + (size_t)(k0 + 32) * N, LDS_OFF(&ldsB[buf ^ 1][0]), 64, 32, (unsigned)N);
    }
#else
    *(v8bf*)&ldsA[buf][arow * 32 + acol]     = ra0;
    *(v8bf*)&ldsA[buf][arow * 32 + acol + 8] = ra1;
    *(v8bf*)&ldsB[buf][brow * 64 + bcol]     = rb0;
    __syncthreads();
    if (k0 + 32 < K) {                    // stage next tile while WMMAs run
      ra0 = *(const v8bf*)(gA + k0 + 32);
      ra1 = *(const v8bf*)(gA + k0 + 40);
      rb0 = *(const v8bf*)(gB + (size_t)(k0 + 32) * N);
      __builtin_prefetch(gA + k0 + 64, 0, 1);        // global_prefetch_b8
      __builtin_prefetch(gB + (size_t)(k0 + 64) * N, 0, 1);
    }
#endif
    v16bf af = frag_a_16x32(&ldsA[buf][(wid * 16) * 32], 32);
#pragma unroll
    for (int tt = 0; tt < 4; ++tt) {
      v16bf bf = frag_b_32x16(&ldsB[buf][0], 64, tt * 16);
      acc[tt] = __builtin_amdgcn_wmma_f32_16x16x32_bf16(
          false, af, false, bf, (short)0, acc[tt], false, false);
    }
  }

  // C layout: VGPR v, lane -> row = v + 8*(lane>>4), col = lane&15
  const int half = lane >> 4, ncol = lane & 15;
  if (outF) {
#pragma unroll
    for (int v = 0; v < 8; ++v) {
      const int row = mBase + wid * 16 + v + 8 * half;
#pragma unroll
      for (int tt = 0; tt < 4; ++tt) {
        const int col = nBase + tt * 16 + ncol;
        outF[(size_t)row * N + col] = acc[tt][v] + bias[col];
      }
    }
  } else {
#pragma unroll
    for (int v = 0; v < 8; ++v) {
      const int row = mBase + wid * 16 + v + 8 * half;
#pragma unroll
      for (int tt = 0; tt < 4; ++tt)
        outBf[(size_t)row * N + nBase + tt * 16 + ncol] = (__bf16)acc[tt][v];
    }
  }
}

// ---------------------------------------------------------------------------
// Flash-style prior @ v. One block per (b,h, 64 i-rows); 4 waves x 16 rows.
// ---------------------------------------------------------------------------
__global__ __launch_bounds__(128) void attn_wmma(const __bf16* __restrict__ Vb,
                                                 const float* __restrict__ Rinv,
                                                 __bf16* __restrict__ Ob) {
  __shared__ __align__(16) __bf16 ldsV[2][32 * 64];
  const int t = threadIdx.x;
  const int wid = t >> 5, lane = t & 31;
  const int bh = blockIdx.y;                 // 0..31
  const int bb = bh >> 3, hh = bh & 7;
  const int iBase = blockIdx.x * 64 + wid * 16;
  const int row = lane & 15, half = lane >> 4;
  const int irow = iBase + row;
  const float ri = Rinv[(size_t)bh * N_ + irow];
  const __bf16* vbase = Vb + (size_t)bb * N_ * INNER_ + hh * DH_;

  v8f acc[4] = {};
  float rsum = 0.0f;

#if USE_TDM
  if (wid == 0) tdm_issue_2d_bf16(vbase, LDS_OFF(&ldsV[0][0]), 64, 32, INNER_);
#else
  const int vr = t >> 2, c0 = (t & 3) * 16;  // 32x64 V tile: 16 elems/thread
  const __bf16* gv = vbase + (size_t)vr * INNER_ + c0;
  v8bf s0 = *(const v8bf*)gv;
  v8bf s1 = *(const v8bf*)(gv + 8);
#endif

  for (int j0 = 0; j0 < N_; j0 += 32) {
    const int buf = (j0 >> 5) & 1;
#if USE_TDM
    if (wid == 0) __builtin_amdgcn_s_wait_tensorcnt(0);  // current tile landed
    __syncthreads();                                     // visible to all waves
    if (wid == 0 && j0 + 32 < N_)                        // prefetch next tile
      tdm_issue_2d_bf16(vbase + (size_t)(j0 + 32) * INNER_,
                        LDS_OFF(&ldsV[buf ^ 1][0]), 64, 32, INNER_);
#else
    *(v8bf*)&ldsV[buf][vr * 64 + c0]     = s0;
    *(v8bf*)&ldsV[buf][vr * 64 + c0 + 8] = s1;
    __syncthreads();
    if (j0 + 32 < N_) {
      s0 = *(const v8bf*)(gv + (size_t)(j0 + 32) * INNER_);
      s1 = *(const v8bf*)(gv + (size_t)(j0 + 32) * INNER_ + 8);
    }
#endif
    v16bf pf;                                 // P tile 16x32 in A-layout
#pragma unroll
    for (int e = 0; e < 8; ++e) {
      const int j  = j0 + half * 8 + e;
      const int j2 = j + 16;
      float p  = __expf(-fabsf((float)(irow - j))  * ri);
      float p2 = __expf(-fabsf((float)(irow - j2)) * ri);
      rsum += p + p2;
      pf[e]     = (__bf16)p;
      pf[e + 8] = (__bf16)p2;
    }
#pragma unroll
    for (int tt = 0; tt < 4; ++tt) {
      v16bf bf = frag_b_32x16(&ldsV[buf][0], 64, tt * 16);
      acc[tt] = __builtin_amdgcn_wmma_f32_16x16x32_bf16(
          false, pf, false, bf, (short)0, acc[tt], false, false);
    }
  }

  rsum += __shfl_xor(rsum, 16, 32);          // lanes l and l+16 share row l&15
#pragma unroll
  for (int v = 0; v < 8; ++v) {
    const float s = __shfl(rsum, v + 8 * half, 32);   // row sum for C row
    const float inv = 1.0f / s;
    const int orow = iBase + v + 8 * half;
    const size_t base = ((size_t)bb * N_ + orow) * INNER_ + hh * DH_;
#pragma unroll
    for (int tt = 0; tt < 4; ++tt)
      Ob[base + tt * 16 + (lane & 15)] = (__bf16)(acc[tt][v] * inv);
  }
}

// ---------------------------------------------------------------------------
extern "C" void kernel_launch(void* const* d_in, const int* in_sizes, int n_in,
                              void* d_out, int out_size, void* d_ws, size_t ws_size,
                              hipStream_t stream) {
  const float* x  = (const float*)d_in[0];
  const float* Wv = (const float*)d_in[1];
  const float* Ws = (const float*)d_in[2];
  const float* bs = (const float*)d_in[3];
  const float* Wo = (const float*)d_in[4];
  const float* bo = (const float*)d_in[5];
  float* out = (float*)d_out;

  // workspace carve-up (~26 MB)
  __bf16* xb  = (__bf16*)d_ws;                       // 8192x512
  __bf16* wvb = xb  + (size_t)M_ * D_;               // 512x512
  __bf16* wob = wvb + (size_t)D_ * INNER_;           // 512x512
  __bf16* vb  = wob + (size_t)INNER_ * D_;           // 8192x512
  __bf16* ob  = vb  + (size_t)M_ * INNER_;           // 8192x512
  float* rinv = (float*)(ob + (size_t)M_ * INNER_);  // 32x2048

  cvt_bf16_kernel<<<(M_ * D_) / 256, 256, 0, stream>>>(x, xb, M_ * D_);
  cvt_bf16_kernel<<<(D_ * INNER_) / 256, 256, 0, stream>>>(Wv, wvb, D_ * INNER_);
  cvt_bf16_kernel<<<(INNER_ * D_) / 256, 256, 0, stream>>>(Wo, wob, INNER_ * D_);

  sigma_kernel<<<M_ / 8, 256, 0, stream>>>(x, Ws, bs, rinv);

  gemm_bf16_wmma<<<dim3(M_ / 128, INNER_ / 64), 256, 0, stream>>>(
      xb, wvb, vb, nullptr, nullptr, M_, INNER_, D_);

  attn_wmma<<<dim3(N_ / 64, B_ * H_), 128, 0, stream>>>(vb, rinv, ob);

  gemm_bf16_wmma<<<dim3(M_ / 128, D_ / 64), 256, 0, stream>>>(
      ob, wob, nullptr, out, bo, M_, D_, INNER_);
}
